// AttentionHead_59425167507842
// MI455X (gfx1250) — matrix operand order repaired and checked by reading, not compiled, not run on hardware
//
#include <hip/hip_runtime.h>

// ---------------------------------------------------------------------------
// Single-head causal attention (B=8, S=2048, D=1024, H=128), fp32 in/out.
// Strategy: f16 WMMA (v_wmma_f32_16x16x32_f16) with f32 accumulation.
//   Kernel 1: q/k/v = x@W + b  -> f16 tensors in workspace.
//   Kernel 2: flash attention, online softmax, O = softmax(QK^T*32)@V.
// Fragment loads are batched ahead of each WMMA chain, with an explicit
// scheduling fence so the backend cannot re-serialize them into
// load+s_wait_dscnt(0)+wmma triplets.
// ---------------------------------------------------------------------------

typedef _Float16 h8   __attribute__((ext_vector_type(8)));
typedef _Float16 v16h __attribute__((ext_vector_type(16)));
typedef float    v8f  __attribute__((ext_vector_type(8)));

union V16 { v16h v; h8 h[2]; };

#define B_ 8
#define S_ 2048
#define D_ 1024
#define H_ 128

#define NEG_BIG (-1.0e30f)

#if __has_builtin(__builtin_amdgcn_sched_barrier)
#define SCHED_FENCE() __builtin_amdgcn_sched_barrier(0)
#else
#define SCHED_FENCE() asm volatile("" ::: "memory")
#endif

static __device__ __forceinline__ v8f wmma_f16(const v16h& a, const v16h& b, const v8f& c) {
  return __builtin_amdgcn_wmma_f32_16x16x32_f16(false, a, false, b, (short)0, c, false, false);
}

// ---------------------------------------------------------------------------
// Kernel 1: QKV projection. grid = (M/128, 3), block = 256 (8 waves).
// Each wave: 16 rows x 128 cols, K = 1024 in chunks of 32.
// ---------------------------------------------------------------------------
__launch_bounds__(256)
__global__ void qkv_proj_kernel(const float* __restrict__ x,
                                const float* __restrict__ Wq, const float* __restrict__ bq,
                                const float* __restrict__ Wk, const float* __restrict__ bk,
                                const float* __restrict__ Wv, const float* __restrict__ bv,
                                _Float16* __restrict__ qkv_ws) {
  const int proj = blockIdx.y;  // 0=q, 1=k, 2=v
  const float* W    = (proj == 0) ? Wq : (proj == 1) ? Wk : Wv;
  const float* bias = (proj == 0) ? bq : (proj == 1) ? bk : bv;
  _Float16* dst = qkv_ws + (size_t)proj * ((size_t)B_ * S_ * H_);

  const int tid   = threadIdx.x;
  const int lane  = tid & 31;
  const int wave  = tid >> 5;
  const int rbase = blockIdx.x * 128 + wave * 16;

  __shared__ __align__(16) _Float16 WldsT[H_ * 32];  // [n][k] transposed W chunk

  v8f acc[8] = {};

  const int arow = rbase + (lane & 15);
  const int aoff = (lane >> 4) * 16;   // A-frag K sub-offset per half-wave
  const int ncol = lane & 15;
  const int koff = (lane >> 4) * 16;   // B-frag K sub-offset per half-wave
  const int half = lane >> 4;

  for (int kc = 0; kc < D_ / 32; ++kc) {
    __syncthreads();
    // Stage W[kc*32 .. kc*32+31][0..127] transposed into LDS as f16.
    {
      const int kk = tid >> 3;            // 0..31
      const int n0 = (tid & 7) * 16;      // 0,16,...,112
      const float* wr = W + (size_t)(kc * 32 + kk) * H_ + n0;
#pragma unroll
      for (int j = 0; j < 16; ++j)
        WldsT[(n0 + j) * 32 + kk] = (_Float16)wr[j];
    }
    __syncthreads();

    // A fragment from x (fp32 -> f16 on the fly).
    V16 a;
    {
      const float* xr = x + (size_t)arow * D_ + kc * 32;
#pragma unroll
      for (int j = 0; j < 8; ++j) a.v[j]     = (_Float16)xr[aoff + j];
#pragma unroll
      for (int j = 0; j < 8; ++j) a.v[8 + j] = (_Float16)xr[8 + aoff + j];
    }

    // Batch all 8 B fragments; fence so the WMMA chain issues back-to-back
    // behind partial dscnt waits instead of one full wait per WMMA.
    V16 bf[8];
#pragma unroll
    for (int nb = 0; nb < 8; ++nb) {
      const _Float16* br = &WldsT[(nb * 16 + ncol) * 32 + koff];
      bf[nb].h[0] = *(const h8*)(br);
      bf[nb].h[1] = *(const h8*)(br + 8);
    }
    SCHED_FENCE();
#pragma unroll
    for (int nb = 0; nb < 8; ++nb)
      acc[nb] = wmma_f16(a.v, bf[nb].v, acc[nb]);
  }

  // Epilogue: bias add, store f16.
#pragma unroll
  for (int nb = 0; nb < 8; ++nb) {
    const float bb = bias[nb * 16 + ncol];
#pragma unroll
    for (int i = 0; i < 8; ++i) {
      const int row = rbase + i + half * 8;
      dst[(size_t)row * H_ + nb * 16 + ncol] = (_Float16)(acc[nb][i] + bb);
    }
  }
}

// ---------------------------------------------------------------------------
// Kernel 2: flash attention. grid = (S/128, B), block = 256 (8 waves).
// Each wave: 16 query rows x full H=128. Key blocks of 32, causal.
// ---------------------------------------------------------------------------
__launch_bounds__(256)
__global__ void flash_attn_kernel(const _Float16* __restrict__ qkv,
                                  float* __restrict__ out) {
  const _Float16* qg = qkv;
  const _Float16* kg = qkv + (size_t)B_ * S_ * H_;
  const _Float16* vg = qkv + (size_t)2 * B_ * S_ * H_;

  const int b    = blockIdx.y;
  const int qb0  = blockIdx.x * 128;
  const int tid  = threadIdx.x;
  const int lane = tid & 31;
  const int wave = tid >> 5;
  const int qb   = qb0 + wave * 16;
  const int qmax = qb + 15;

  __shared__ __align__(16) _Float16 Klds[32 * H_];      // K block, row-major
  __shared__ __align__(16) _Float16 VldsT[H_ * 32];     // V block, transposed [n][k]
  __shared__ __align__(16) _Float16 Plds[8][16 * 32];   // per-wave P tile

  const int arow = lane & 15;
  const int aoff = (lane >> 4) * 16;
  const int ncol = lane & 15;
  const int koff = (lane >> 4) * 16;
  const int half = lane >> 4;

  // Preload Q fragments (16 rows x 128 K) into registers, 4 chunks of 32.
  V16 qf[4];
#pragma unroll
  for (int kc = 0; kc < 4; ++kc) {
    const _Float16* p = qg + ((size_t)b * S_ + qb + arow) * H_ + kc * 32;
    qf[kc].h[0] = *(const h8*)(p + aoff);
    qf[kc].h[1] = *(const h8*)(p + 8 + aoff);
  }

  v8f accO[8] = {};
  float mrow[8], lrow[8];
#pragma unroll
  for (int i = 0; i < 8; ++i) { mrow[i] = NEG_BIG; lrow[i] = 0.0f; }

  const int nblocks = (qb0 + 128) / 32;

  for (int kb = 0; kb < nblocks; ++kb) {
    const int kbase = kb * 32;
    __syncthreads();
    // Cooperative staging: K row-major, V transposed. 256 threads, 32B each.
    {
      const int r  = tid >> 3;            // 0..31
      const int c0 = (tid & 7) * 16;      // 0..112
      const _Float16* ksrc = kg + ((size_t)b * S_ + kbase + r) * H_ + c0;
      *(h8*)&Klds[r * H_ + c0]     = *(const h8*)(ksrc);
      *(h8*)&Klds[r * H_ + c0 + 8] = *(const h8*)(ksrc + 8);
      const _Float16* vsrc = vg + ((size_t)b * S_ + kbase + r) * H_ + c0;
#pragma unroll
      for (int j = 0; j < 16; ++j)
        VldsT[(c0 + j) * 32 + r] = vsrc[j];
      if (kb + 1 < nblocks) {  // prefetch next K/V block -> global_prefetch
        __builtin_prefetch(kg + ((size_t)b * S_ + kbase + 32 + r) * H_ + c0, 0, 0);
        __builtin_prefetch(vg + ((size_t)b * S_ + kbase + 32 + r) * H_ + c0, 0, 0);
      }
    }
    __syncthreads();

    if (kbase <= qmax) {
      // ---- Batch-load all 8 K fragments for the 16x32 score tile.
      V16 kf[8];
#pragma unroll
      for (int kc = 0; kc < 4; ++kc) {
        const _Float16* kr0 = &Klds[(ncol)      * H_ + kc * 32 + koff];
        const _Float16* kr1 = &Klds[(16 + ncol) * H_ + kc * 32 + koff];
        kf[2 * kc].h[0]     = *(const h8*)(kr0);
        kf[2 * kc].h[1]     = *(const h8*)(kr0 + 8);
        kf[2 * kc + 1].h[0] = *(const h8*)(kr1);
        kf[2 * kc + 1].h[1] = *(const h8*)(kr1 + 8);
      }
      SCHED_FENCE();

      // ---- S = (Q @ K^T): two 16x16 accumulators, 8 chained WMMAs.
      v8f s0 = {}, s1 = {};
#pragma unroll
      for (int kc = 0; kc < 4; ++kc) {
        s0 = wmma_f16(qf[kc].v, kf[2 * kc].v,     s0);
        s1 = wmma_f16(qf[kc].v, kf[2 * kc + 1].v, s1);
      }

      // ---- Prefetch all 8 V fragments now; fence pins their issue point
      //      here so LDS latency hides behind the softmax VALU work below.
      V16 vf[8];
#pragma unroll
      for (int nb = 0; nb < 8; ++nb) {
        const _Float16* vr = &VldsT[(nb * 16 + ncol) * 32 + koff];
        vf[nb].h[0] = *(const h8*)(vr);
        vf[nb].h[1] = *(const h8*)(vr + 8);
      }
      SCHED_FENCE();

      // ---- scale (reference multiplies by sqrt(D)=32) + causal mask.
      const float scale = 32.0f;
#pragma unroll
      for (int i = 0; i < 8; ++i) {
        const int row = qb + i + half * 8;
        float e0 = s0[i] * scale;
        float e1 = s1[i] * scale;
        if (kbase + ncol      > row) e0 = NEG_BIG;
        if (kbase + 16 + ncol > row) e1 = NEG_BIG;
        s0[i] = e0; s1[i] = e1;
      }

      // ---- online softmax update per row slot.
#pragma unroll
      for (int i = 0; i < 8; ++i) {
        float mb = fmaxf(s0[i], s1[i]);
        mb = fmaxf(mb, __shfl_xor(mb, 1, 32));
        mb = fmaxf(mb, __shfl_xor(mb, 2, 32));
        mb = fmaxf(mb, __shfl_xor(mb, 4, 32));
        mb = fmaxf(mb, __shfl_xor(mb, 8, 32));
        const float mn   = fmaxf(mrow[i], mb);
        const float corr = __expf(mrow[i] - mn);
        mrow[i] = mn;
        const float p0 = __expf(s0[i] - mn);
        const float p1 = __expf(s1[i] - mn);
        float rs = p0 + p1;
        rs += __shfl_xor(rs, 1, 32);
        rs += __shfl_xor(rs, 2, 32);
        rs += __shfl_xor(rs, 4, 32);
        rs += __shfl_xor(rs, 8, 32);
        lrow[i] = lrow[i] * corr + rs;
#pragma unroll
        for (int nb = 0; nb < 8; ++nb) accO[nb][i] *= corr;
        // park P (f16) in per-wave LDS tile for C-layout -> A-layout reshape
        Plds[wave][(i + half * 8) * 32 + ncol]      = (_Float16)p0;
        Plds[wave][(i + half * 8) * 32 + 16 + ncol] = (_Float16)p1;
      }

      // DS ops are in-order per wave, but make the store->load edge explicit.
      asm volatile("s_wait_dscnt 0" ::: "memory");

      // ---- O += P @ V : A = P (16x32), B = V fragments (already resident).
      V16 pf;
      const _Float16* pr = &Plds[wave][arow * 32];
      pf.h[0] = *(const h8*)(pr + aoff);
      pf.h[1] = *(const h8*)(pr + 8 + aoff);
#pragma unroll
      for (int nb = 0; nb < 8; ++nb)
        accO[nb] = wmma_f16(pf.v, vf[nb].v, accO[nb]);
    }
  }

  // ---- normalize and store fp32 output.
#pragma unroll
  for (int i = 0; i < 8; ++i) {
    const float inv = 1.0f / lrow[i];
    const int row = qb + i + half * 8;
    float* orow = out + ((size_t)b * S_ + row) * H_;
#pragma unroll
    for (int nb = 0; nb < 8; ++nb)
      orow[nb * 16 + ncol] = accO[nb][i] * inv;
  }
}

// ---------------------------------------------------------------------------
extern "C" void kernel_launch(void* const* d_in, const int* in_sizes, int n_in,
                              void* d_out, int out_size, void* d_ws, size_t ws_size,
                              hipStream_t stream) {
  (void)in_sizes; (void)n_in; (void)out_size; (void)ws_size;
  const float* x  = (const float*)d_in[0];
  const float* Wq = (const float*)d_in[1];
  const float* bq = (const float*)d_in[2];
  const float* Wk = (const float*)d_in[3];
  const float* bk = (const float*)d_in[4];
  const float* Wv = (const float*)d_in[5];
  const float* bv = (const float*)d_in[6];
  float* out = (float*)d_out;

  // workspace: q,k,v as f16, each B*S*H elements (12 MB total)
  _Float16* qkv = (_Float16*)d_ws;

  dim3 g1((B_ * S_) / 128, 3);
  qkv_proj_kernel<<<g1, 256, 0, stream>>>(x, Wq, bq, Wk, bk, Wv, bv, qkv);

  dim3 g2(S_ / 128, B_);
  flash_attn_kernel<<<g2, 256, 0, stream>>>(qkv, out);
}
